// RelationalSymbolicAttention_22832046145744
// MI455X (gfx1250) — matrix operand order repaired and checked by reading, not compiled, not run on hardware
//
#include <hip/hip_runtime.h>

#define BB   4
#define SS   4096
#define DD   1024
#define HH   8
#define JJ   16
#define DH   128
#define NSYM 512
#define MM   (BB * SS)   /* 16384 rows */

typedef __attribute__((ext_vector_type(16))) __bf16 v16bf;
typedef __attribute__((ext_vector_type(8)))  float  v8f;

// ---------------------------------------------------------------------------
// Prep: transpose fp32 (R x C) -> bf16 (C x R)
// ---------------------------------------------------------------------------
__global__ void k_transpose_bf16(const float* __restrict__ in, __bf16* __restrict__ outT,
                                 int R, int C) {
  long i = (long)blockIdx.x * blockDim.x + threadIdx.x;
  if (i >= (long)R * C) return;
  int r = (int)(i / C), c = (int)(i % C);
  outT[(long)c * R + r] = (__bf16)in[(long)r * C + c];
}

__global__ void k_convert_bf16(const float* __restrict__ in, __bf16* __restrict__ out, long n) {
  long i = (long)blockIdx.x * blockDim.x + threadIdx.x;
  if (i < n) out[i] = (__bf16)in[i];
}

// ---------------------------------------------------------------------------
// WMMA GEMM: C_bf16[M,N] = A[M,K] @ BT(N,K)^T + bias[N],  N == 1024.
// Block = 256 threads (8 waves) owns a 16 x 1024 output stripe.
// Per k-step the block stages the shared 16x32 A tile in LDS (converted to
// bf16 once), then each wave builds its A fragment from LDS and runs 8 WMMAs
// over its 16x128 column slice.  grid = (M/16).
// ---------------------------------------------------------------------------
template <typename TA>
__global__ __launch_bounds__(256)
void k_gemm_wmma(const TA* __restrict__ A, const __bf16* __restrict__ BT,
                 const float* __restrict__ bias, __bf16* __restrict__ C,
                 int M, int N, int K) {
  __shared__ __bf16 aLds[16 * 32];
  int lane = threadIdx.x & 31;
  int wave = threadIdx.x >> 5;
  int row0 = blockIdx.x * 16;
  int col0 = wave * 128;
  int ln = lane & 15;
  int hi = lane >> 4;  // 0 or 1

  v8f acc[8] = {};
  for (int k0 = 0; k0 < K; k0 += 32) {
    __syncthreads();  // previous iteration's LDS reads done
    // cooperative stage+convert of A[row0..row0+15][k0..k0+31] -> LDS
#pragma unroll
    for (int u = 0; u < 2; ++u) {
      int e = threadIdx.x * 2 + u;      // 0..511
      int r = e >> 5, k = e & 31;
      aLds[r * 32 + k] = (__bf16)(float)A[(long)(row0 + r) * K + k0 + k];
    }
    __syncthreads();
    // A fragment: contiguous k order (k = 16*hi + j, same as B) gathered from LDS
    v16bf a;
#pragma unroll
    for (int j = 0; j < 16; ++j) {
      int ka = j + hi * 16;
      a[j] = aLds[ln * 32 + ka];
    }
#pragma unroll
    for (int t = 0; t < 8; ++t) {
      int n = col0 + t * 16 + ln;
      const __bf16* bp = BT + (long)n * K + k0 + hi * 16;  // contiguous in K
      v16bf b;
#pragma unroll
      for (int j = 0; j < 16; ++j) b[j] = bp[j];
      acc[t] = __builtin_amdgcn_wmma_f32_16x16x32_bf16(
          false, a, false, b, (short)0, acc[t], false, false);
    }
  }
#pragma unroll
  for (int t = 0; t < 8; ++t) {
    int n = col0 + t * 16 + ln;
    float bv = bias[n];
#pragma unroll
    for (int r = 0; r < 8; ++r) {
      int row = row0 + r + hi * 8;
      C[(long)row * N + n] = (__bf16)(acc[t][r] + bv);
    }
  }
}

// ---------------------------------------------------------------------------
// Banded relative attention: for each (b,h,s) dot q against 16 previous keys,
// softmax over j, write probs to (b, s, j*H + h) layout in bf16.
// ---------------------------------------------------------------------------
__global__ __launch_bounds__(256)
void k_rel_attn(const __bf16* __restrict__ qb, const __bf16* __restrict__ kb,
                __bf16* __restrict__ relp) {
  long gid = (long)blockIdx.x * blockDim.x + threadIdx.x;
  if (gid >= (long)BB * HH * SS) return;
  int b = (int)(gid / (HH * SS));
  int rem = (int)(gid % (HH * SS));
  int h = rem / SS;
  int s = rem % SS;
  const __bf16* qrow = qb + ((long)(b * SS + s) * DD + h * DH);
  const __bf16* kr[JJ];
#pragma unroll
  for (int j = 0; j < JJ; ++j) {
    int sj = s - (j + 1); if (sj < 0) sj = 0;
    kr[j] = kb + ((long)(b * SS + sj) * DD + h * DH);
  }
  float dots[JJ];
#pragma unroll
  for (int j = 0; j < JJ; ++j) dots[j] = 0.f;
  for (int e = 0; e < DH; ++e) {
    float qv = (float)qrow[e];
#pragma unroll
    for (int j = 0; j < JJ; ++j) dots[j] += qv * (float)kr[j][e];
  }
  const float scale = 0.08838834764831845f;  // 1/sqrt(128)
  float mx = -1e30f;
#pragma unroll
  for (int j = 0; j < JJ; ++j) { dots[j] *= scale; mx = fmaxf(mx, dots[j]); }
  float sum = 0.f;
#pragma unroll
  for (int j = 0; j < JJ; ++j) { dots[j] = __expf(dots[j] - mx); sum += dots[j]; }
  float inv = 1.f / sum;
  __bf16* orow = relp + (long)(b * SS + s) * (JJ * HH);
#pragma unroll
  for (int j = 0; j < JJ; ++j) orow[j * HH + h] = (__bf16)(dots[j] * inv);
}

// ---------------------------------------------------------------------------
// Fused symbolic attention (flash-style, all-WMMA):
// per wave: 16 query rows x 1 head; stream 512 symbols in chunks of 32.
// scores -> online softmax -> P@V, accumulators rescaled in registers.
// grid = (M/16/8, H), block = 256 (8 waves).
// ---------------------------------------------------------------------------
__global__ __launch_bounds__(256)
void k_sym_attn(const __bf16* __restrict__ q2, const __bf16* __restrict__ tmpl,
                const __bf16* __restrict__ symT, float* __restrict__ out) {
  __shared__ __bf16 plds[8][16 * 32];
  int lane = threadIdx.x & 31;
  int wave = threadIdx.x >> 5;
  int h = blockIdx.y;
  int row0 = (blockIdx.x * 8 + wave) * 16;
  int ln = lane & 15;
  int hi = lane >> 4;

  // A fragments of q2 head slice (K = 128 -> 4 fragments), kept resident
  v16bf af[4];
#pragma unroll
  for (int kk = 0; kk < 4; ++kk) {
#pragma unroll
    for (int j = 0; j < 16; ++j) {
      int ka = h * DH + kk * 32 + j + hi * 16;
      af[kk][j] = q2[(long)(row0 + ln) * DD + ka];
    }
  }

  v8f oacc[8] = {};
  float mrun[8], lrun[8];
#pragma unroll
  for (int r = 0; r < 8; ++r) { mrun[r] = -1e30f; lrun[r] = 0.f; }
  const float scale2 = 0.08838834764831845f;  // 1/sqrt(128)

  for (int c = 0; c < NSYM / 32; ++c) {
    // scores: two 16x16 subtiles over a 32-symbol chunk, K=128 in 4 steps
    v8f s0 = {}, s1 = {};
#pragma unroll
    for (int kk = 0; kk < 4; ++kk) {
      const __bf16* b0p = tmpl + (long)(c * 32 + ln) * DD + h * DH + kk * 32 + hi * 16;
      const __bf16* b1p = b0p + 16 * DD;
      v16bf b0, b1;
#pragma unroll
      for (int j = 0; j < 16; ++j) { b0[j] = b0p[j]; b1[j] = b1p[j]; }
      s0 = __builtin_amdgcn_wmma_f32_16x16x32_bf16(false, af[kk], false, b0, (short)0, s0, false, false);
      s1 = __builtin_amdgcn_wmma_f32_16x16x32_bf16(false, af[kk], false, b1, (short)0, s1, false, false);
    }
    // online softmax: row stats live across the 16 lanes of each half-wave
#pragma unroll
    for (int r = 0; r < 8; ++r) {
      float v0 = s0[r] * scale2;
      float v1 = s1[r] * scale2;
      float rm = fmaxf(v0, v1);
      rm = fmaxf(rm, __shfl_xor(rm, 1, 32));
      rm = fmaxf(rm, __shfl_xor(rm, 2, 32));
      rm = fmaxf(rm, __shfl_xor(rm, 4, 32));
      rm = fmaxf(rm, __shfl_xor(rm, 8, 32));
      float nm = fmaxf(mrun[r], rm);
      float p0 = __expf(v0 - nm);
      float p1 = __expf(v1 - nm);
      float rs = p0 + p1;
      rs += __shfl_xor(rs, 1, 32);
      rs += __shfl_xor(rs, 2, 32);
      rs += __shfl_xor(rs, 4, 32);
      rs += __shfl_xor(rs, 8, 32);
      float alpha = __expf(mrun[r] - nm);
      lrun[r] = lrun[r] * alpha + rs;
      mrun[r] = nm;
#pragma unroll
      for (int t = 0; t < 8; ++t) oacc[t][r] *= alpha;
      int prow = r + hi * 8;  // C-layout row owned by this lane/VGPR
      plds[wave][prow * 32 + ln]      = (__bf16)p0;
      plds[wave][prow * 32 + 16 + ln] = (__bf16)p1;
    }
    __syncthreads();
    // re-read probs as a 16x32 A fragment
    v16bf pa;
#pragma unroll
    for (int j = 0; j < 16; ++j) {
      int ka = j + hi * 16;
      pa[j] = plds[wave][ln * 32 + ka];
    }
    __syncthreads();
    // out += P @ V_chunk : 8 output column tiles of dh=128
#pragma unroll
    for (int t = 0; t < 8; ++t) {
      const __bf16* vp = symT + (long)(h * DH + t * 16 + ln) * NSYM + c * 32 + hi * 16;
      v16bf vb;
#pragma unroll
      for (int j = 0; j < 16; ++j) vb[j] = vp[j];
      oacc[t] = __builtin_amdgcn_wmma_f32_16x16x32_bf16(
          false, pa, false, vb, (short)0, oacc[t], false, false);
    }
  }
  // normalize and store fp32 output (B,S,d) with column h*128..+127
#pragma unroll
  for (int t = 0; t < 8; ++t) {
    int n = h * DH + t * 16 + ln;
#pragma unroll
    for (int r = 0; r < 8; ++r) {
      int row = row0 + r + hi * 8;
      out[(long)row * DD + n] = oacc[t][r] / lrun[r];
    }
  }
}

// ---------------------------------------------------------------------------
extern "C" void kernel_launch(void* const* d_in, const int* in_sizes, int n_in,
                              void* d_out, int out_size, void* d_ws, size_t ws_size,
                              hipStream_t stream) {
  const float* x     = (const float*)d_in[0];
  const float* Wq    = (const float*)d_in[1];
  const float* bq    = (const float*)d_in[2];
  const float* Wk    = (const float*)d_in[3];
  const float* bk    = (const float*)d_in[4];
  const float* Wp    = (const float*)d_in[5];
  const float* bp    = (const float*)d_in[6];
  const float* Wq2   = (const float*)d_in[7];
  const float* bq2   = (const float*)d_in[8];
  const float* tmplf = (const float*)d_in[9];
  const float* symf  = (const float*)d_in[10];
  float* out = (float*)d_out;

  char* ws = (char*)d_ws;
  size_t off = 0;
  auto wsalloc = [&](size_t bytes) -> char* {
    char* p = ws + off;
    off += (bytes + 255) & ~(size_t)255;
    return p;
  };
  __bf16* WqT   = (__bf16*)wsalloc((size_t)DD * DD * 2);
  __bf16* WkT   = (__bf16*)wsalloc((size_t)DD * DD * 2);
  __bf16* Wq2T  = (__bf16*)wsalloc((size_t)DD * DD * 2);
  __bf16* WpT   = (__bf16*)wsalloc((size_t)DD * DH * 2);
  __bf16* tmplb = (__bf16*)wsalloc((size_t)NSYM * DD * 2);
  __bf16* symT  = (__bf16*)wsalloc((size_t)DD * NSYM * 2);
  __bf16* qb    = (__bf16*)wsalloc((size_t)MM * DD * 2);
  __bf16* kb    = (__bf16*)wsalloc((size_t)MM * DD * 2);
  __bf16* relp  = (__bf16*)wsalloc((size_t)MM * (JJ * HH) * 2);
  __bf16* yb    = (__bf16*)wsalloc((size_t)MM * DD * 2);
  __bf16* q2b   = (__bf16*)wsalloc((size_t)MM * DD * 2);

  const int TPB = 256;
  // weight prep
  k_transpose_bf16<<<((long)DD * DD + TPB - 1) / TPB, TPB, 0, stream>>>(Wq, WqT, DD, DD);
  k_transpose_bf16<<<((long)DD * DD + TPB - 1) / TPB, TPB, 0, stream>>>(Wk, WkT, DD, DD);
  k_transpose_bf16<<<((long)DD * DD + TPB - 1) / TPB, TPB, 0, stream>>>(Wq2, Wq2T, DD, DD);
  k_transpose_bf16<<<((long)DH * DD + TPB - 1) / TPB, TPB, 0, stream>>>(Wp, WpT, JJ * HH, DD);
  k_convert_bf16<<<((long)NSYM * DD + TPB - 1) / TPB, TPB, 0, stream>>>(tmplf, tmplb, (long)NSYM * DD);
  k_transpose_bf16<<<((long)NSYM * DD + TPB - 1) / TPB, TPB, 0, stream>>>(symf, symT, NSYM, DD);

  // q, k projections (WMMA)
  dim3 ggrid(MM / 16);
  k_gemm_wmma<float><<<ggrid, TPB, 0, stream>>>(x, WqT, bq, qb, MM, DD, DD);
  k_gemm_wmma<float><<<ggrid, TPB, 0, stream>>>(x, WkT, bk, kb, MM, DD, DD);

  // banded relative attention + softmax
  k_rel_attn<<<((long)BB * HH * SS + TPB - 1) / TPB, TPB, 0, stream>>>(qb, kb, relp);

  // y = rel @ Wp + bp ; q2 = y @ Wq2 + bq2 (WMMA)
  k_gemm_wmma<__bf16><<<ggrid, TPB, 0, stream>>>(relp, WpT, bp, yb, MM, DD, JJ * HH);
  k_gemm_wmma<__bf16><<<ggrid, TPB, 0, stream>>>(yb, Wq2T, bq2, q2b, MM, DD, DD);

  // fused symbolic attention (WMMA, flash-style)
  k_sym_attn<<<dim3(MM / 16 / 8, HH), TPB, 0, stream>>>(q2b, tmplb, symT, out);
}
